// Discriminator_42391327211890
// MI455X (gfx1250) — matrix operand order repaired and checked by reading, not compiled, not run on hardware
//
#include <hip/hip_runtime.h>
#include <hip/hip_bf16.h>

// ---------------------------------------------------------------------------
// Decomposable-attention NLI forward for MI455X (gfx1250, wave32, WMMA).
// All GEMMs: v_wmma_f32_16x16x32_bf16, 128x64 block tile, 8 waves, each wave
// a 32x32 tile (4 accumulators). LDS double-buffered; tiles staged with
// GLOBAL_LOAD_ASYNC_TO_LDS_B128 (ASYNCcnt) when the toolchain declares the
// builtin, sync b128 staging otherwise. B operands are always bf16 [N,K]
// (weights natively; activation-side B pre-transposed once).
// ---------------------------------------------------------------------------

typedef __attribute__((ext_vector_type(16))) __bf16          v16bf;
typedef __attribute__((ext_vector_type(16))) unsigned short  v16u;
typedef __attribute__((ext_vector_type(8)))  unsigned short  u16x8;
typedef __attribute__((ext_vector_type(8)))  float           v8f;
typedef __attribute__((ext_vector_type(4)))  float           f32x4;
typedef __attribute__((ext_vector_type(4)))  int             v4i;
typedef __attribute__((address_space(1))) v4i*               as1_v4i;
typedef __attribute__((address_space(3))) v4i*               as3_v4i;

#define LSEQ 4096      // LP == LH
#define D0   300       // embedding dim
#define DP   320       // padded embedding dim (mult of 32)
#define HD   512       // hidden
#define D2   600       // 2*D
#define D2P  608       // padded 2*D (mult of 32)

__device__ __forceinline__ unsigned short f2bf(float f) {
  unsigned int u = __float_as_uint(f);
  u += 0x7FFFu + ((u >> 16) & 1u);           // round-to-nearest-even
  return (unsigned short)(u >> 16);
}

// ---- gfx1250 async global->LDS copy (ASYNCcnt) with safe fallback ---------
#ifndef HAVE_ASYNC_LDS
#if defined(__has_builtin)
#if __has_builtin(__builtin_amdgcn_global_load_async_to_lds_b128)
#define HAVE_ASYNC_LDS 1
#endif
#endif
#endif
#ifndef HAVE_ASYNC_LDS
#define HAVE_ASYNC_LDS 0
#endif

#if HAVE_ASYNC_LDS
__device__ __forceinline__ void async_b128(const unsigned short* g, unsigned short* l) {
  __builtin_amdgcn_global_load_async_to_lds_b128(
      (as1_v4i)(v4i*)(void*)const_cast<unsigned short*>(g),
      (as3_v4i)(v4i*)(void*)l, 0, 0);
}
template <int N> __device__ __forceinline__ void wait_async() {
#if __has_builtin(__builtin_amdgcn_s_wait_asynccnt)
  __builtin_amdgcn_s_wait_asynccnt(N);
#else
  asm volatile("s_wait_asynccnt %0" ::"i"(N) : "memory");
#endif
}
#else
__device__ __forceinline__ void async_b128(const unsigned short* g, unsigned short* l) {
  *(u16x8*)l = *(const u16x8*)g;             // sync fallback
}
template <int N> __device__ __forceinline__ void wait_async() {}
#endif

// ---------------------------------------------------------------------------
// WMMA GEMM:  C = act(A @ Bt^T + bias),  Bt is bf16 [N,K] (contiguous K).
//   A modes: bf16 row-major [M,K] | f32 row-major scaled per row |
//            f32 transposed (A' = E^T) scaled per A'-row (per E column).
// Requires M%128==0, K%32==0, lda/ldb%8==0; N guarded at stores.
// ---------------------------------------------------------------------------
enum AMode { A_BF16 = 0, A_F32_SCALED = 1, A_F32_SCALED_T = 2 };

template <int AM, bool RELU_BIAS, bool OUTF, bool OUTB>
__global__ __launch_bounds__(256) void gemm_wmma_kernel(
    const void* __restrict__ Aptr, const float* __restrict__ Ascale, int lda,
    const unsigned short* __restrict__ Bt, int ldb,
    const float* __restrict__ bias,
    float* __restrict__ Cf, unsigned short* __restrict__ Cb, int ldc,
    int M, int N, int K)
{
  // row stride 40 ushorts = 80B: every 8-element offset is 16B aligned
  __shared__ __align__(16) unsigned short sA[2][128][40];   // [buf][m][k]
  __shared__ __align__(16) unsigned short sB[2][64][40];    // [buf][n][k]

  const int tid  = threadIdx.x;
  const int lane = tid & 31;
  const int wave = tid >> 5;
  const int half = (lane >> 4) & 1;
  const int l16  = lane & 15;
  const int wm   = wave & 3;      // 32-row slab
  const int wn   = wave >> 2;     // 32-col slab
  const int bm   = blockIdx.y * 128;
  const int bn   = blockIdx.x * 64;

  v8f acc[2][2] = {};

  // staging indices (constant across K loop)
  const int ar  = tid >> 1;            // A row 0..127
  const int ac  = (tid & 1) * 16;      // A col chunk (16 elems)
  const int ti  = tid >> 3;            // trans: E-row-local 0..31
  const int tj  = (tid & 7) * 16;      // trans: E-col chunk 0..112
  const int bn4 = tid >> 2;            // B row (n) 0..63
  const int bc  = (tid & 3) * 8;       // B col chunk (8 elems)

  float rowScale = 0.f;
  float colScale[16];
  if constexpr (AM == A_F32_SCALED) rowScale = Ascale[bm + ar];
  if constexpr (AM == A_F32_SCALED_T) {
#pragma unroll
    for (int q = 0; q < 16; ++q) colScale[q] = Ascale[bm + tj + q];
  }

  auto stage = [&](int k0, int buf) {
    if constexpr (AM == A_BF16) {
      const unsigned short* src =
          (const unsigned short*)Aptr + (size_t)(bm + ar) * lda + k0 + ac;
      async_b128(src,     &sA[buf][ar][ac]);
      async_b128(src + 8, &sA[buf][ar][ac + 8]);
    } else if constexpr (AM == A_F32_SCALED) {
      const float* s = (const float*)Aptr + (size_t)(bm + ar) * lda + k0 + ac;
#pragma unroll
      for (int q4 = 0; q4 < 4; ++q4) {
        f32x4 v = *(const f32x4*)(s + q4 * 4);
#pragma unroll
        for (int q = 0; q < 4; ++q)
          sA[buf][ar][ac + q4 * 4 + q] = f2bf(v[q] * rowScale);
      }
    } else {  // A' = E^T: read E[k0+ti][bm+tj..+15], store transposed+scaled
      const float* s = (const float*)Aptr + (size_t)(k0 + ti) * lda + bm + tj;
#pragma unroll
      for (int q4 = 0; q4 < 4; ++q4) {
        f32x4 v = *(const f32x4*)(s + q4 * 4);
#pragma unroll
        for (int q = 0; q < 4; ++q)
          sA[buf][tj + q4 * 4 + q][ti] = f2bf(v[q] * colScale[q4 * 4 + q]);
      }
    }
    async_b128(Bt + (size_t)(bn + bn4) * ldb + k0 + bc, &sB[buf][bn4][bc]);
  };

  constexpr int INFLIGHT = (AM == A_BF16) ? 3 : 1;  // async ops of newest stage

  const int T = K >> 5;
  stage(0, 0);
  for (int t = 0; t < T; ++t) {
    const int cur = t & 1;
    if (t + 1 < T) { stage(t + 1, cur ^ 1); wait_async<INFLIGHT>(); }
    else           { wait_async<0>(); }
    __syncthreads();

    // fragments per ISA bf16 A(16x32)/B(32x16) layouts: 2x ds_load_b128 each
    v16bf afrag[2];
#pragma unroll
    for (int sm = 0; sm < 2; ++sm) {
      const int arow = wm * 32 + sm * 16 + l16;
      u16x8 lo = *(const u16x8*)&sA[cur][arow][half * 8];
      u16x8 hi = *(const u16x8*)&sA[cur][arow][16 + half * 8];
      v16u u;
#pragma unroll
      for (int q = 0; q < 8; ++q) { u[q] = lo[q]; u[8 + q] = hi[q]; }
      afrag[sm] = __builtin_bit_cast(v16bf, u);
    }
#pragma unroll
    for (int sn = 0; sn < 2; ++sn) {
      const int bcol = wn * 32 + sn * 16 + l16;
      u16x8 lo = *(const u16x8*)&sB[cur][bcol][half * 8];
      u16x8 hi = *(const u16x8*)&sB[cur][bcol][16 + half * 8];
      v16u u;
#pragma unroll
      for (int q = 0; q < 8; ++q) { u[q] = lo[q]; u[8 + q] = hi[q]; }
      v16bf bfrag = __builtin_bit_cast(v16bf, u);
#pragma unroll
      for (int sm = 0; sm < 2; ++sm)
        acc[sm][sn] = __builtin_amdgcn_wmma_f32_16x16x32_bf16(
            false, afrag[sm], false, bfrag, (short)0, acc[sm][sn], false, false);
    }
    __syncthreads();
  }

  // epilogue: D lanes<16: M=r, N=lane; lanes>=16: M=r+8, N=lane-16
#pragma unroll
  for (int sm = 0; sm < 2; ++sm)
#pragma unroll
    for (int sn = 0; sn < 2; ++sn) {
      const int gn = bn + wn * 32 + sn * 16 + l16;
#pragma unroll
      for (int r = 0; r < 8; ++r) {
        const int gm = bm + wm * 32 + sm * 16 + half * 8 + r;
        if (gm < M && gn < N) {
          float v = acc[sm][sn][r];
          if constexpr (RELU_BIAS) { v += bias[gn]; v = v > 0.f ? v : 0.f; }
          if constexpr (OUTF) Cf[(size_t)gm * ldc + gn] = v;
          if constexpr (OUTB) Cb[(size_t)gm * ldc + gn] = f2bf(v);
        }
      }
    }
}

// ---------------------------------------------------------------------------
// Helper kernels
// ---------------------------------------------------------------------------
__global__ __launch_bounds__(256) void transpose_bf16_kernel(
    const unsigned short* __restrict__ in, unsigned short* __restrict__ out,
    int R, int C) {  // out[C][R] = in[R][C]^T
  __shared__ unsigned short tile[32][33];
  const int tx = threadIdx.x & 31, ty = threadIdx.x >> 5;
  const int c0 = blockIdx.x * 32, r0 = blockIdx.y * 32;
#pragma unroll
  for (int p = 0; p < 4; ++p) {
    int r = r0 + ty + p * 8, c = c0 + tx;
    tile[ty + p * 8][tx] = (r < R && c < C) ? in[(size_t)r * C + c] : (unsigned short)0;
  }
  __syncthreads();
#pragma unroll
  for (int p = 0; p < 4; ++p) {
    int orow = c0 + ty + p * 8, oc = r0 + tx;
    if (orow < C && oc < R) out[(size_t)orow * R + oc] = tile[tx][ty + p * 8];
  }
}

__global__ void gather_bf16_kernel(const int* __restrict__ idx,
                                   const float* __restrict__ emb,
                                   unsigned short* __restrict__ out, int L) {
  int g = blockIdx.x * blockDim.x + threadIdx.x;
  if (g >= L * DP) return;
  int i = g / DP, d = g - i * DP;
  float v = (d < D0) ? emb[(size_t)idx[i] * D0 + d] : 0.f;
  out[g] = f2bf(v);
}

__global__ void convert_pad_kernel(const float* __restrict__ src,
                                   unsigned short* __restrict__ dst,
                                   int N, int K, int ldd) {
  int g = blockIdx.x * blockDim.x + threadIdx.x;
  if (g >= N * ldd) return;
  int n = g / ldd, k = g - n * ldd;
  dst[g] = (k < K) ? f2bf(src[(size_t)n * K + k]) : (unsigned short)0;
}

__global__ void concat_kernel(const unsigned short* __restrict__ embb,  // [L,DP]
                              const float* __restrict__ soft,           // [L,D0]
                              unsigned short* __restrict__ cat, int L) {
  int g = blockIdx.x * blockDim.x + threadIdx.x;
  if (g >= L * D2P) return;
  int i = g / D2P, k = g - i * D2P;
  unsigned short v;
  if (k < D0)      v = embb[(size_t)i * DP + k];
  else if (k < D2) v = f2bf(soft[(size_t)i * D0 + (k - D0)]);
  else             v = 0;
  cat[g] = v;
}

__global__ void rowsum_recip_kernel(const float* __restrict__ E,
                                    float* __restrict__ out, int N) {
  __shared__ float red[256];
  const float* p = E + (size_t)blockIdx.x * N;
  float s = 0.f;
  for (int j = threadIdx.x; j < N; j += 256) s += p[j];
  red[threadIdx.x] = s;
  __syncthreads();
  for (int off = 128; off > 0; off >>= 1) {
    if (threadIdx.x < off) red[threadIdx.x] += red[threadIdx.x + off];
    __syncthreads();
  }
  if (threadIdx.x == 0) out[blockIdx.x] = 1.0f / red[0];
}

__global__ void colsum_recip_kernel(const float* __restrict__ E,
                                    float* __restrict__ out, int M, int N) {
  int j = blockIdx.x * blockDim.x + threadIdx.x;
  if (j >= N) return;
  float s = 0.f;
  for (int i = 0; i < M; ++i) s += E[(size_t)i * N + j];
  out[j] = 1.0f / s;
}

__global__ void colsum_kernel(const float* __restrict__ X,
                              float* __restrict__ out, int M, int N) {
  int j = blockIdx.x * blockDim.x + threadIdx.x;
  if (j >= N) return;
  float s = 0.f;
  for (int i = 0; i < M; ++i) s += X[(size_t)i * N + j];
  out[j] = s;
}

__global__ __launch_bounds__(1024) void head_kernel(
    const float* __restrict__ v1, const float* __restrict__ v2,
    const float* __restrict__ Wg1, const float* __restrict__ bg1,
    const float* __restrict__ Wg2, const float* __restrict__ bg2,
    const float* __restrict__ Wg3, const float* __restrict__ bg3,
    float* __restrict__ y) {
  __shared__ float cat[1024], y1[512], y2[512], z[3];
  int t = threadIdx.x;
  cat[t] = (t < 512) ? v1[t] : v2[t - 512];
  __syncthreads();
  if (t < 512) {
    float s = bg1[t];
    const float* w = Wg1 + (size_t)t * 1024;
    for (int k = 0; k < 1024; ++k) s += w[k] * cat[k];
    y1[t] = s > 0.f ? s : 0.f;
  }
  __syncthreads();
  if (t < 512) {
    float s = bg2[t];
    const float* w = Wg2 + (size_t)t * 512;
    for (int k = 0; k < 512; ++k) s += w[k] * y1[k];
    y2[t] = s > 0.f ? s : 0.f;
  }
  __syncthreads();
  if (t < 3) {
    float s = bg3[t];
    const float* w = Wg3 + (size_t)t * 512;
    for (int k = 0; k < 512; ++k) s += w[k] * y2[k];
    z[t] = s;
  }
  __syncthreads();
  if (t == 0) {
    float m  = fmaxf(z[0], fmaxf(z[1], z[2]));
    float e0 = __expf(z[0] - m), e1 = __expf(z[1] - m), e2 = __expf(z[2] - m);
    float inv = 1.f / (e0 + e1 + e2);
    y[0] = e0 * inv; y[1] = e1 * inv; y[2] = e2 * inv;
  }
}

// ---------------------------------------------------------------------------
extern "C" void kernel_launch(void* const* d_in, const int* in_sizes, int n_in,
                              void* d_out, int out_size, void* d_ws, size_t ws_size,
                              hipStream_t stream) {
  (void)in_sizes; (void)n_in; (void)out_size; (void)ws_size;
  const int*   p_idx = (const int*)d_in[0];
  const int*   h_idx = (const int*)d_in[1];
  const float* emb   = (const float*)d_in[2];
  const float* W_a1  = (const float*)d_in[3];
  const float* b_a1  = (const float*)d_in[4];
  const float* W_a2  = (const float*)d_in[5];
  const float* b_a2  = (const float*)d_in[6];
  const float* W_c1  = (const float*)d_in[7];
  const float* b_c1  = (const float*)d_in[8];
  const float* W_c2  = (const float*)d_in[9];
  const float* b_c2  = (const float*)d_in[10];
  const float* W_g1  = (const float*)d_in[11];
  const float* b_g1  = (const float*)d_in[12];
  const float* W_g2  = (const float*)d_in[13];
  const float* b_g2  = (const float*)d_in[14];
  const float* W_g3  = (const float*)d_in[15];
  const float* b_g3  = (const float*)d_in[16];

  float* out   = (float*)d_out;
  float* E     = out;                                 // [4096,4096]
  float* beta  = E + (size_t)LSEQ * LSEQ;             // [4096,300]
  float* alpha = beta + (size_t)LSEQ * D0;            // [4096,300]
  float* v1    = alpha + (size_t)LSEQ * D0;           // [512]
  float* v2    = v1 + HD;                             // [512]
  float* y     = v2 + HD;                             // [3]

  char* w = (char*)d_ws;
  auto alloc = [&](size_t bytes) -> char* {
    char* p = w; w += (bytes + 255) & ~(size_t)255; return p;
  };
  unsigned short* pbuf   = (unsigned short*)alloc((size_t)LSEQ * DP * 2);
  unsigned short* hbuf   = (unsigned short*)alloc((size_t)LSEQ * DP * 2);
  unsigned short* pbufT  = (unsigned short*)alloc((size_t)DP * LSEQ * 2);
  unsigned short* hbufT  = (unsigned short*)alloc((size_t)DP * LSEQ * 2);
  unsigned short* Wa1b   = (unsigned short*)alloc((size_t)HD * DP * 2);
  unsigned short* Wa2b   = (unsigned short*)alloc((size_t)HD * HD * 2);
  unsigned short* Wc1b   = (unsigned short*)alloc((size_t)HD * D2P * 2);
  unsigned short* Wc2b   = (unsigned short*)alloc((size_t)HD * HD * 2);
  unsigned short* fp1    = (unsigned short*)alloc((size_t)LSEQ * HD * 2);
  unsigned short* fh1    = (unsigned short*)alloc((size_t)LSEQ * HD * 2);
  unsigned short* fp     = (unsigned short*)alloc((size_t)LSEQ * HD * 2);
  unsigned short* fh     = (unsigned short*)alloc((size_t)LSEQ * HD * 2);
  unsigned short* fhT    = (unsigned short*)alloc((size_t)LSEQ * HD * 2);
  float*          r_eik  = (float*)alloc((size_t)LSEQ * 4);
  float*          r_ekj  = (float*)alloc((size_t)LSEQ * 4);
  unsigned short* catP   = (unsigned short*)alloc((size_t)LSEQ * D2P * 2);
  unsigned short* catH   = (unsigned short*)alloc((size_t)LSEQ * D2P * 2);
  unsigned short* comp1P = (unsigned short*)alloc((size_t)LSEQ * HD * 2);
  unsigned short* comp1H = (unsigned short*)alloc((size_t)LSEQ * HD * 2);
  float*          comp2P = (float*)alloc((size_t)LSEQ * HD * 4);
  float*          comp2H = (float*)alloc((size_t)LSEQ * HD * 4);

  // --- precision conversion / gathers / activation transposes ---
  convert_pad_kernel<<<(HD * DP + 255) / 256, 256, 0, stream>>>(W_a1, Wa1b, HD, D0, DP);
  convert_pad_kernel<<<(HD * HD + 255) / 256, 256, 0, stream>>>(W_a2, Wa2b, HD, HD, HD);
  convert_pad_kernel<<<(HD * D2P + 255) / 256, 256, 0, stream>>>(W_c1, Wc1b, HD, D2, D2P);
  convert_pad_kernel<<<(HD * HD + 255) / 256, 256, 0, stream>>>(W_c2, Wc2b, HD, HD, HD);
  gather_bf16_kernel<<<(LSEQ * DP + 255) / 256, 256, 0, stream>>>(p_idx, emb, pbuf, LSEQ);
  gather_bf16_kernel<<<(LSEQ * DP + 255) / 256, 256, 0, stream>>>(h_idx, emb, hbuf, LSEQ);
  transpose_bf16_kernel<<<dim3(DP / 32, LSEQ / 32), 256, 0, stream>>>(pbuf, pbufT, LSEQ, DP);
  transpose_bf16_kernel<<<dim3(DP / 32, LSEQ / 32), 256, 0, stream>>>(hbuf, hbufT, LSEQ, DP);

  const dim3 gH(HD / 64, LSEQ / 128);     // N=512 tiles
  const dim3 gE(LSEQ / 64, LSEQ / 128);   // N=4096 tiles
  const dim3 gD((D0 + 63) / 64, LSEQ / 128);

  // --- attend MLP ---
  gemm_wmma_kernel<A_BF16, true, false, true><<<gH, 256, 0, stream>>>(
      pbuf, nullptr, DP, Wa1b, DP, b_a1, nullptr, fp1, HD, LSEQ, HD, DP);
  gemm_wmma_kernel<A_BF16, true, false, true><<<gH, 256, 0, stream>>>(
      hbuf, nullptr, DP, Wa1b, DP, b_a1, nullptr, fh1, HD, LSEQ, HD, DP);
  gemm_wmma_kernel<A_BF16, true, false, true><<<gH, 256, 0, stream>>>(
      fp1, nullptr, HD, Wa2b, HD, b_a2, nullptr, fp, HD, LSEQ, HD, HD);
  gemm_wmma_kernel<A_BF16, true, false, true><<<gH, 256, 0, stream>>>(
      fh1, nullptr, HD, Wa2b, HD, b_a2, nullptr, fh, HD, LSEQ, HD, HD);

  // --- E = fp @ fh.reshape(512,4096): Bt = reshape^T = [4096,512] ---
  transpose_bf16_kernel<<<dim3(LSEQ / 32, HD / 32), 256, 0, stream>>>(fh, fhT, HD, LSEQ);
  gemm_wmma_kernel<A_BF16, false, true, false><<<gE, 256, 0, stream>>>(
      fp, nullptr, HD, fhT, HD, nullptr, E, nullptr, LSEQ, LSEQ, LSEQ, HD);

  // --- row / column reciprocal sums ---
  rowsum_recip_kernel<<<LSEQ, 256, 0, stream>>>(E, r_eik, LSEQ);
  colsum_recip_kernel<<<LSEQ / 256, 256, 0, stream>>>(E, r_ekj, LSEQ, LSEQ);

  // --- beta = (E/eik) @ h_emb ; alpha = (E/ekj)^T @ p_emb ---
  gemm_wmma_kernel<A_F32_SCALED, false, true, false><<<gD, 256, 0, stream>>>(
      E, r_eik, LSEQ, hbufT, LSEQ, nullptr, beta, nullptr, D0, LSEQ, D0, LSEQ);
  gemm_wmma_kernel<A_F32_SCALED_T, false, true, false><<<gD, 256, 0, stream>>>(
      E, r_ekj, LSEQ, pbufT, LSEQ, nullptr, alpha, nullptr, D0, LSEQ, D0, LSEQ);

  // --- compare MLP on concat(emb, soft-align) ---
  concat_kernel<<<(LSEQ * D2P + 255) / 256, 256, 0, stream>>>(pbuf, beta, catP, LSEQ);
  concat_kernel<<<(LSEQ * D2P + 255) / 256, 256, 0, stream>>>(hbuf, alpha, catH, LSEQ);
  gemm_wmma_kernel<A_BF16, true, false, true><<<gH, 256, 0, stream>>>(
      catP, nullptr, D2P, Wc1b, D2P, b_c1, nullptr, comp1P, HD, LSEQ, HD, D2P);
  gemm_wmma_kernel<A_BF16, true, false, true><<<gH, 256, 0, stream>>>(
      catH, nullptr, D2P, Wc1b, D2P, b_c1, nullptr, comp1H, HD, LSEQ, HD, D2P);
  gemm_wmma_kernel<A_BF16, true, true, false><<<gH, 256, 0, stream>>>(
      comp1P, nullptr, HD, Wc2b, HD, b_c2, comp2P, nullptr, HD, LSEQ, HD, HD);
  gemm_wmma_kernel<A_BF16, true, true, false><<<gH, 256, 0, stream>>>(
      comp1H, nullptr, HD, Wc2b, HD, b_c2, comp2H, nullptr, HD, LSEQ, HD, HD);

  // --- v1 / v2 (sum over rows) and classifier head ---
  colsum_kernel<<<HD / 256, 256, 0, stream>>>(comp2P, v1, LSEQ, HD);
  colsum_kernel<<<HD / 256, 256, 0, stream>>>(comp2H, v2, LSEQ, HD);
  head_kernel<<<1, 1024, 0, stream>>>(v1, v2, W_g1, b_g1, W_g2, b_g2, W_g3, b_g3, y);
}
